// DIN_3100966387950
// MI455X (gfx1250) — matrix operand order repaired and checked by reading, not compiled, not run on hardware
//
#include <hip/hip_runtime.h>
#include <hip/hip_bf16.h>

typedef __attribute__((ext_vector_type(16))) _Float16 v16h;
typedef __attribute__((ext_vector_type(8)))  _Float16 v8h;
typedef __attribute__((ext_vector_type(2)))  _Float16 v2h;
typedef __attribute__((ext_vector_type(8)))  float    v8f;

#define B_   4096
#define T_   200
#define H_   64
#define R_   (B_*T_)     /* 819200 rows of the activation unit */
#define K4H  256
#define AUH  36
#define AUP  48          /* N padded to 3 WMMA tiles */

/* ---- stats block layout (floats at base of workspace) ---- */
#define S_AU_SUM   0
#define S_AU_SQ    48
#define S_AU_MEAN  96
#define S_AU_SCL   144
#define S_M1_SUM   192
#define S_M1_SQ    272
#define S_M1_MEAN  352
#define S_M1_SCL   432
#define S_M2_SUM   512
#define S_M2_SQ    552
#define S_M2_MEAN  592
#define S_M2_SCL   632
#define STATS_FLOATS 1024

/* ---- workspace layout ---- */
#define WS_W1T_BYTE 4096                       /* f16 [48][256] = 24KB      */
#define WS_W_OFF    7168                       /* w[B*T] floats             */
#define WS_CUR_OFF  (WS_W_OFF + R_)            /* cur[B*64]                 */
#define WS_Y1_OFF   (WS_CUR_OFF + B_*H_)       /* y1[B*80]                  */
#define WS_Y2_OFF   (WS_Y1_OFF + B_*80)        /* y2[B*40]                  */
#define WS_H1_BYTE  ((size_t)(WS_Y2_OFF + B_*40) * 4)  /* h1T f16 [36][R]   */

__global__ void zero_stats_kernel(float* __restrict__ S) {
    int i = threadIdx.x + blockIdx.x * blockDim.x;
    if (i < STATS_FLOATS) S[i] = 0.0f;
}

/* Precompute au_W1 transposed to [n][k], f16, zero-padded n -> 48. */
__global__ void prep_w1_kernel(const float* __restrict__ au_W1,
                               _Float16* __restrict__ w1t) {
    int i = blockIdx.x * 256 + threadIdx.x;
    if (i < AUP * K4H) {
        int n = i >> 8, k = i & 255;
        w1t[i] = (_Float16)((n < AUH) ? au_W1[k * AUH + n] : 0.0f);
    }
}

/* ============================================================
 * Kernel A: h1_pre = concat(he, ie, he-ie, he*ie) @ au_W1 + b1
 * WMMA f16->f32; weights pulled to LDS via async-to-LDS b128;
 * K-chunk outer / N-tile inner: each A fragment loaded once and
 * fed to 3 consecutive v_wmma (3 live accumulators);
 * output stored transposed (h1T[ch][r]) as b128 per lane;
 * per-channel sum / sumsq accumulated via LDS + global atomics.
 * Block = 128 threads (4 waves), 64 rows/block, 12800 blocks.
 * ============================================================ */
__global__ __launch_bounds__(128) void au_gemm1_kernel(
    const int* __restrict__ hist, const int* __restrict__ item,
    const float* __restrict__ Itab, const _Float16* __restrict__ w1t,
    const float* __restrict__ au_b1,
    _Float16* __restrict__ h1t, float* __restrict__ S)
{
    __shared__ _Float16 sW1[AUP * K4H];   /* [n][k] transposed, 24KB */
    __shared__ _Float16 sA [64 * K4H];    /* [row][k] au_in tile,   32KB */
    __shared__ float    sStat[2 * AUP];

    const int tid  = threadIdx.x;
    const int wave = tid >> 5;
    const int lane = tid & 31;

    if (tid < 2 * AUP) sStat[tid] = 0.0f;

    /* async copy of the prepped f16 weight image: 24KB = 128 thr * 12 * 16B */
    {
        const uint32_t ldsBase = (uint32_t)(uintptr_t)(&sW1[0]);
        #pragma unroll
        for (int it = 0; it < 12; ++it) {
            uint32_t boff = (uint32_t)((it * 128 + tid) * 16);
            uint32_t ldsAddr = ldsBase + boff;
            asm volatile("global_load_async_to_lds_b128 %0, %1, %2"
                         :: "v"(ldsAddr), "v"(boff), "s"(w1t) : "memory");
        }
    }

    /* build au_in rows in f16: [he | ie | he-ie | he*ie] */
    const int rowBase = blockIdx.x * 64;
    #pragma unroll 1
    for (int m = 0; m < 16; ++m) {
        int rl = wave * 16 + m;
        int r  = rowBase + rl;
        int b  = r / T_;
        const float2* hep = (const float2*)(Itab + (size_t)hist[r] * H_);
        const float2* iep = (const float2*)(Itab + (size_t)item[b] * H_);
        float2 hv = hep[lane];
        float2 iv = iep[lane];
        _Float16* dst = sA + rl * K4H + 2 * lane;
        *(v2h*)(dst)       = (v2h){(_Float16)hv.x,            (_Float16)hv.y};
        *(v2h*)(dst + 64)  = (v2h){(_Float16)iv.x,            (_Float16)iv.y};
        *(v2h*)(dst + 128) = (v2h){(_Float16)(hv.x - iv.x),   (_Float16)(hv.y - iv.y)};
        *(v2h*)(dst + 192) = (v2h){(_Float16)(hv.x * iv.x),   (_Float16)(hv.y * iv.y)};
    }
    asm volatile("s_wait_asynccnt 0" ::: "memory");
    __syncthreads();

    /* each wave: one 16-row M tile; K = 8 chunks of 32; 3 N tiles inner */
    const int mlo  = lane & 15;
    const int hsel = lane >> 4;
    const _Float16* aRow = sA + (wave * 16 + mlo) * K4H;
    const _Float16* bCol = sW1 + mlo * K4H;   /* + nt*16*K4H per N tile */

    v8f acc[3];
    #pragma unroll
    for (int nt = 0; nt < 3; ++nt)
        acc[nt] = (v8f){0.f, 0.f, 0.f, 0.f, 0.f, 0.f, 0.f, 0.f};

    #pragma unroll
    for (int c = 0; c < 8; ++c) {
        /* A: lanes 0-15 hold K {0..7,16..23}; lanes 16-31 hold K {8..15,24..31} */
        const int ka = 32 * c + hsel * 8;
        v8h a0 = *(const v8h*)(aRow + ka);
        v8h a1 = *(const v8h*)(aRow + ka + 16);
        v16h af;
        #pragma unroll
        for (int i = 0; i < 8; ++i) { af[i] = a0[i]; af[8 + i] = a1[i]; }

        /* B: lanes 0-15 hold K {0..7,16..23}; lanes 16-31 hold K {8..15,24..31} (col = lane&15) */
        const int kb = 32 * c + hsel * 8;
        #pragma unroll
        for (int nt = 0; nt < 3; ++nt) {
            const _Float16* bp = bCol + nt * 16 * K4H + kb;
            v8h b0  = *(const v8h*)(bp);
            v8h b1v = *(const v8h*)(bp + 16);
            v16h bf;
            #pragma unroll
            for (int i = 0; i < 8; ++i) { bf[i] = b0[i]; bf[8 + i] = b1v[i]; }
            acc[nt] = __builtin_amdgcn_wmma_f32_16x16x32_f16(
                          false, af, false, bf, (short)0, acc[nt], false, false);
        }
    }

    #pragma unroll
    for (int nt = 0; nt < 3; ++nt) {
        const int ch = nt * 16 + mlo;
        const float bias = (ch < AUH) ? au_b1[ch] : 0.0f;
        float s = 0.0f, sq = 0.0f;
        v8h hv;
        #pragma unroll
        for (int i = 0; i < 8; ++i) {
            float v = acc[nt][i] + bias;   /* row M = hsel*8+i, col = ch */
            s += v; sq += v * v;
            hv[i] = (_Float16)v;
        }
        if (ch < AUH) {
            /* contiguous 8 rows for this channel -> single b128 store */
            *(v8h*)(h1t + (size_t)ch * R_ + rowBase + wave * 16 + hsel * 8) = hv;
        }
        atomicAdd(&sStat[ch], s);
        atomicAdd(&sStat[AUP + ch], sq);
    }
    __syncthreads();
    if (tid < AUH)                          atomicAdd(&S[S_AU_SUM + tid], sStat[tid]);
    else if (tid >= AUP && tid < AUP + AUH) atomicAdd(&S[S_AU_SQ + (tid - AUP)], sStat[tid]);
}

/* mean = sum/N ; scale = rsqrt(var + eps) */
__global__ void finalize_kernel(float* __restrict__ S, int sumOff, int sqOff,
                                int meanOff, int sclOff, int n, float invCount) {
    int i = threadIdx.x + blockIdx.x * blockDim.x;
    if (i < n) {
        float m = S[sumOff + i] * invCount;
        float v = S[sqOff + i] * invCount - m * m;
        S[meanOff + i] = m;
        S[sclOff + i]  = rsqrtf(v + 1e-8f);
    }
}

/* Kernel C1: w[r] = dice(h1[r]) . au_W2 + au_b2  (h1 transposed layout) */
__global__ __launch_bounds__(256) void au_dice_gemm2_kernel(
    const _Float16* __restrict__ h1t, const float* __restrict__ S,
    const float* __restrict__ au_a1, const float* __restrict__ au_W2,
    const float* __restrict__ au_b2, float* __restrict__ w)
{
    int r = blockIdx.x * 256 + threadIdx.x;
    float acc = au_b2[0];
    #pragma unroll
    for (int j = 0; j < AUH; ++j) {
        float h  = (float)h1t[(size_t)j * R_ + r];   /* coalesced across threads */
        float xn = (h - S[S_AU_MEAN + j]) * S[S_AU_SCL + j];
        float p  = 1.0f / (1.0f + __expf(-xn));
        float a  = au_a1[j];
        acc += h * (p * (1.0f - a) + a) * au_W2[j];
    }
    w[r] = acc;
}

/* Kernel C2: cur[b] = sum_t w[b,t] * he[b,t]   (one wave per batch row) */
__global__ __launch_bounds__(256) void pool_kernel(
    const float* __restrict__ w, const int* __restrict__ hist,
    const float* __restrict__ Itab, float* __restrict__ cur)
{
    int wave = threadIdx.x >> 5, lane = threadIdx.x & 31;
    int b = blockIdx.x * 8 + wave;
    float2 acc = {0.0f, 0.0f};
    int base = b * T_;
    #pragma unroll 4
    for (int t = 0; t < T_; ++t) {
        int r = base + t;
        float wv = w[r];
        const float2* hp = (const float2*)(Itab + (size_t)hist[r] * H_);
        float2 hv = hp[lane];
        acc.x += wv * hv.x;
        acc.y += wv * hv.y;
    }
    ((float2*)(cur + (size_t)b * H_))[lane] = acc;
}

/* Kernel D: y1 = [ue|ie|ce|cur] @ W1 + b1, stats(80). Block per row. */
__global__ __launch_bounds__(128) void mlp1_kernel(
    const int* __restrict__ user, const int* __restrict__ item,
    const int* __restrict__ cate,
    const float* __restrict__ Utab, const float* __restrict__ Itab,
    const float* __restrict__ Ctab, const float* __restrict__ cur,
    const float* __restrict__ W1, const float* __restrict__ b1,
    float* __restrict__ y1, float* __restrict__ S)
{
    __shared__ float sx[K4H];
    int b = blockIdx.x, tid = threadIdx.x;
    for (int i = tid; i < K4H; i += 128) {
        int seg = i >> 6, off = i & 63;
        float v;
        if      (seg == 0) v = Utab[(size_t)user[b] * H_ + off];
        else if (seg == 1) v = Itab[(size_t)item[b] * H_ + off];
        else if (seg == 2) v = Ctab[(size_t)cate[b] * H_ + off];
        else               v = cur[(size_t)b * H_ + off];
        sx[i] = v;
    }
    __syncthreads();
    if (tid < 80) {
        float acc = b1[tid];
        for (int k = 0; k < K4H; ++k) acc += sx[k] * W1[k * 80 + tid];
        y1[(size_t)b * 80 + tid] = acc;
        atomicAdd(&S[S_M1_SUM + tid], acc);
        atomicAdd(&S[S_M1_SQ  + tid], acc * acc);
    }
}

/* Kernel E: y2 = dice(y1) @ W2 + b2, stats(40). Block per row. */
__global__ __launch_bounds__(128) void mlp2_kernel(
    const float* __restrict__ y1, float* __restrict__ S,
    const float* __restrict__ a1, const float* __restrict__ W2,
    const float* __restrict__ b2, float* __restrict__ y2)
{
    __shared__ float sy[80];
    int b = blockIdx.x, tid = threadIdx.x;
    if (tid < 80) {
        float h  = y1[(size_t)b * 80 + tid];
        float xn = (h - S[S_M1_MEAN + tid]) * S[S_M1_SCL + tid];
        float p  = 1.0f / (1.0f + __expf(-xn));
        float a  = a1[tid];
        sy[tid]  = h * (p * (1.0f - a) + a);
    }
    __syncthreads();
    if (tid < 40) {
        float acc = b2[tid];
        for (int k = 0; k < 80; ++k) acc += sy[k] * W2[k * 40 + tid];
        y2[(size_t)b * 40 + tid] = acc;
        atomicAdd(&S[S_M2_SUM + tid], acc);
        atomicAdd(&S[S_M2_SQ  + tid], acc * acc);
    }
}

/* Kernel F: out = dice(y2) @ W3 + b3. Thread per row. */
__global__ __launch_bounds__(256) void mlp3_kernel(
    const float* __restrict__ y2, const float* __restrict__ S,
    const float* __restrict__ a2, const float* __restrict__ W3,
    const float* __restrict__ b3, float* __restrict__ out)
{
    int b = blockIdx.x * 256 + threadIdx.x;
    if (b >= B_) return;
    float acc0 = b3[0], acc1 = b3[1];
    for (int k = 0; k < 40; ++k) {
        float h  = y2[(size_t)b * 40 + k];
        float xn = (h - S[S_M2_MEAN + k]) * S[S_M2_SCL + k];
        float p  = 1.0f / (1.0f + __expf(-xn));
        float a  = a2[k];
        float d  = h * (p * (1.0f - a) + a);
        acc0 += d * W3[k * 2 + 0];
        acc1 += d * W3[k * 2 + 1];
    }
    out[b * 2 + 0] = acc0;
    out[b * 2 + 1] = acc1;
}

extern "C" void kernel_launch(void* const* d_in, const int* in_sizes, int n_in,
                              void* d_out, int out_size, void* d_ws, size_t ws_size,
                              hipStream_t stream)
{
    (void)in_sizes; (void)n_in; (void)out_size; (void)ws_size;
    const int*   user  = (const int*)  d_in[0];
    const int*   hist  = (const int*)  d_in[1];
    const int*   item  = (const int*)  d_in[2];
    const int*   cate  = (const int*)  d_in[3];
    const float* Utab  = (const float*)d_in[4];
    const float* Itab  = (const float*)d_in[5];
    const float* Ctab  = (const float*)d_in[6];
    const float* au_W1 = (const float*)d_in[7];
    const float* au_b1 = (const float*)d_in[8];
    const float* au_a1 = (const float*)d_in[9];
    const float* au_W2 = (const float*)d_in[10];
    const float* au_b2 = (const float*)d_in[11];
    const float* W1    = (const float*)d_in[12];
    const float* b1    = (const float*)d_in[13];
    const float* a1    = (const float*)d_in[14];
    const float* W2    = (const float*)d_in[15];
    const float* b2    = (const float*)d_in[16];
    const float* a2    = (const float*)d_in[17];
    const float* W3    = (const float*)d_in[18];
    const float* b3    = (const float*)d_in[19];
    float* out = (float*)d_out;

    float* S   = (float*)d_ws;
    _Float16* w1t = (_Float16*)((char*)d_ws + WS_W1T_BYTE);
    float* w   = S + WS_W_OFF;
    float* cur = S + WS_CUR_OFF;
    float* y1  = S + WS_Y1_OFF;
    float* y2  = S + WS_Y2_OFF;
    _Float16* h1t = (_Float16*)((char*)d_ws + WS_H1_BYTE);

    zero_stats_kernel<<<4, 256, 0, stream>>>(S);
    prep_w1_kernel<<<48, 256, 0, stream>>>(au_W1, w1t);

    au_gemm1_kernel<<<R_ / 64, 128, 0, stream>>>(hist, item, Itab, w1t, au_b1, h1t, S);
    finalize_kernel<<<1, 64, 0, stream>>>(S, S_AU_SUM, S_AU_SQ, S_AU_MEAN, S_AU_SCL,
                                          AUH, 1.0f / (float)R_);
    au_dice_gemm2_kernel<<<R_ / 256, 256, 0, stream>>>(h1t, S, au_a1, au_W2, au_b2, w);
    pool_kernel<<<B_ / 8, 256, 0, stream>>>(w, hist, Itab, cur);

    mlp1_kernel<<<B_, 128, 0, stream>>>(user, item, cate, Utab, Itab, Ctab,
                                        cur, W1, b1, y1, S);
    finalize_kernel<<<1, 128, 0, stream>>>(S, S_M1_SUM, S_M1_SQ, S_M1_MEAN, S_M1_SCL,
                                           80, 1.0f / (float)B_);
    mlp2_kernel<<<B_, 128, 0, stream>>>(y1, S, a1, W2, b2, y2);
    finalize_kernel<<<1, 64, 0, stream>>>(S, S_M2_SUM, S_M2_SQ, S_M2_MEAN, S_M2_SCL,
                                          40, 1.0f / (float)B_);
    mlp3_kernel<<<B_ / 256, 256, 0, stream>>>(y2, S, a2, W3, b3, out);
}
